// EventDenoisingMamba_83193516523958
// MI455X (gfx1250) — matrix operand order repaired and checked by reading, not compiled, not run on hardware
//
#include <hip/hip_runtime.h>
#include <hip/hip_bf16.h>

// ---------------------------------------------------------------------------
// EventDenoisingMamba for MI455X (gfx1250, wave32, WMMA).
//   GEMMs: bf16 WMMA 16x16x32, f32 accumulate. Block tile 128x128, 8 waves,
//          each wave 32x64 (2x4 WMMA tiles).
//   Scan:  chunked parallel scan (64 chunks x 64 steps), elementwise-linear
//          recurrence composed associatively -> critical path 4096 -> ~192.
// Requires ws_size >= ~116 MB.
// ---------------------------------------------------------------------------

#define B_  2
#define L_  4096
#define F_  4
#define DM_ 256
#define DI_ 512
#define DS_ 16
#define DC_ 4
#define NL_ 4
#define DR_ 16
#define ML_ (B_ * L_)          // 8192 rows in flattened activation matrices
#define PROJW_ (DR_ + 2 * DS_) // 48
#define CH_  64                // scan chunk length
#define NCH_ (L_ / CH_)        // 64 chunks

typedef __attribute__((ext_vector_type(16))) __bf16 v16bf;
typedef __attribute__((ext_vector_type(8)))  __bf16 v8bf;
typedef __attribute__((ext_vector_type(8)))  float  v8f;

__device__ __forceinline__ float silu_f(float x) { return x / (1.0f + __expf(-x)); }

// ---------------------------------------------------------------------------
// embed: x[m, dm] = dot(features[m, :4], embed_W[dm, :4]) + embed_b[dm]
// ---------------------------------------------------------------------------
__global__ __launch_bounds__(256) void embed_kernel(
    const float* __restrict__ f, const float* __restrict__ Wd,
    const float* __restrict__ bd, float* __restrict__ x)
{
    int idx = blockIdx.x * 256 + threadIdx.x;
    if (idx >= ML_ * DM_) return;
    int m = idx / DM_, dm = idx % DM_;
    const float* fr = f + (size_t)m * F_;
    const float* wr = Wd + (size_t)dm * F_;
    x[idx] = fr[0]*wr[0] + fr[1]*wr[1] + fr[2]*wr[2] + fr[3]*wr[3] + bd[dm];
}

// ---------------------------------------------------------------------------
// f32 -> bf16 staging
// ---------------------------------------------------------------------------
__global__ __launch_bounds__(256) void cvt_kernel(
    const float* __restrict__ in, __bf16* __restrict__ out, int n)
{
    int i = blockIdx.x * 256 + threadIdx.x;
    if (i < n) out[i] = (__bf16)in[i];
}

// ---------------------------------------------------------------------------
// WMMA GEMM: C[M,N] (f32) = A[M,K] (bf16, row-major) @ W[N,K]^T (bf16).
// Block tile 128x128, 8 waves (4x2); wave computes 32x64 = 2x4 WMMA tiles.
// M, K multiples of 128/32; N arbitrary (masked).
// ---------------------------------------------------------------------------
__global__ __launch_bounds__(256) void gemm_wmma_bf16(
    const __bf16* __restrict__ A, const __bf16* __restrict__ W,
    float* __restrict__ C, int M, int N, int K)
{
    __shared__ __bf16 As[128 * 40];  // 128 rows x 32 halfs, pad stride 40
    __shared__ __bf16 Ws[128 * 40];  // 128 rows (N) x 32 halfs

    const int tid  = threadIdx.x;
    const int lane = tid & 31;
    const int wv   = tid >> 5;
    const int wm   = wv >> 1;        // 0..3 -> 32-row strips
    const int wn   = wv & 1;         // 0..1 -> 64-col strips
    const int m0   = blockIdx.y * 128;
    const int n0   = blockIdx.x * 128;

    v8f acc[2][4] = {};

    for (int k0 = 0; k0 < K; k0 += 32) {
        // --- cooperative loads into LDS (16B vector loads) ---
        #pragma unroll
        for (int i = 0; i < 2; ++i) {
            int idx = tid + i * 256;          // 512 chunks of 8 halfs
            int r = idx >> 2, cc = idx & 3;
            uint4 v = *(const uint4*)(A + (size_t)(m0 + r) * K + k0 + cc * 8);
            *(uint4*)(As + r * 40 + cc * 8) = v;
        }
        #pragma unroll
        for (int i = 0; i < 2; ++i) {
            int idx = tid + i * 256;          // 512 chunks of 8 halfs
            int r = idx >> 2, cc = idx & 3;
            uint4 v = make_uint4(0u, 0u, 0u, 0u);
            if (n0 + r < N)
                v = *(const uint4*)(W + (size_t)(n0 + r) * K + k0 + cc * 8);
            *(uint4*)(Ws + r * 40 + cc * 8) = v;
        }
        // speculative prefetch of next K-slab of A (global_prefetch_b8)
        if (k0 + 32 < K)
            __builtin_prefetch(A + (size_t)(m0 + (tid >> 2)) * K + (k0 + 32), 0, 0);
        __syncthreads();

        // --- build fragments (ISA 7.12.2 / 7.12.4 layouts) ---
        const int ml = lane & 15;
        const int kh = lane >> 4;     // which K-half this lane carries
        v16bf afrag[2], bfrag[4];
        #pragma unroll
        for (int i = 0; i < 2; ++i) {
            // A lane L: M = L&15; L<16: K {0..7,16..23}; L>=16: K {8..15,24..31}
            const __bf16* ar = As + (wm * 32 + i * 16 + ml) * 40;
            v8bf lo = *(const v8bf*)(ar + kh * 8);
            v8bf hi = *(const v8bf*)(ar + 16 + kh * 8);
            afrag[i] = __builtin_shufflevector(lo, hi,
                0,1,2,3,4,5,6,7,8,9,10,11,12,13,14,15);
        }
        #pragma unroll
        for (int j = 0; j < 4; ++j) {
            // B lane L: N = L&15; L<16: K {0..7,16..23}; L>=16: K {8..15,24..31}
            const __bf16* br = Ws + (wn * 64 + j * 16 + ml) * 40;
            v8bf lo = *(const v8bf*)(br + kh * 8);
            v8bf hi = *(const v8bf*)(br + 16 + kh * 8);
            bfrag[j] = __builtin_shufflevector(lo, hi,
                0,1,2,3,4,5,6,7,8,9,10,11,12,13,14,15);
        }

        #pragma unroll
        for (int i = 0; i < 2; ++i)
            #pragma unroll
            for (int j = 0; j < 4; ++j)
                acc[i][j] = __builtin_amdgcn_wmma_f32_16x16x32_bf16(
                    false, afrag[i], false, bfrag[j],
                    (short)0, acc[i][j], false, false);
        __syncthreads();
    }

    // --- store C: VGPR r -> (M = r + 8*(lane>>4), N = lane&15) ---
    const int nl = lane & 15;
    const int mh = lane >> 4;
    #pragma unroll
    for (int i = 0; i < 2; ++i) {
        #pragma unroll
        for (int j = 0; j < 4; ++j) {
            int ncol = n0 + wn * 64 + j * 16 + nl;
            if (ncol < N) {
                #pragma unroll
                for (int r = 0; r < 8; ++r) {
                    int mrow = m0 + wm * 32 + i * 16 + r + mh * 8;
                    C[(size_t)mrow * N + ncol] = acc[i][j][r];
                }
            }
        }
    }
}

// ---------------------------------------------------------------------------
// depthwise causal conv (kernel 4) + bias + silu.  Input = xh half of xz.
// ---------------------------------------------------------------------------
__global__ __launch_bounds__(256) void conv_silu_kernel(
    const float* __restrict__ xz, const float* __restrict__ Wc,
    const float* __restrict__ bc, float* __restrict__ xh)
{
    int idx = blockIdx.x * 256 + threadIdx.x;
    if (idx >= ML_ * DI_) return;
    int m = idx / DI_, d = idx % DI_;
    int t = m % L_;
    float acc = bc[d];
    #pragma unroll
    for (int j = 0; j < DC_; ++j) {
        int tt = t - (DC_ - 1) + j;
        if (tt >= 0)
            acc += xz[(size_t)(m - (DC_ - 1) + j) * (2 * DI_) + d] * Wc[d * DC_ + j];
    }
    xh[idx] = silu_f(acc);
}

// ---------------------------------------------------------------------------
// dt = softplus(proj[:, :16] @ Wdt^T + bdt)   (K=16 -> scalar FMAs)
// ---------------------------------------------------------------------------
__global__ __launch_bounds__(256) void dt_kernel(
    const float* __restrict__ proj, const float* __restrict__ Wdt,
    const float* __restrict__ bdt, float* __restrict__ dt)
{
    int idx = blockIdx.x * 256 + threadIdx.x;
    if (idx >= ML_ * DI_) return;
    int m = idx / DI_, d = idx % DI_;
    const float* pr = proj + (size_t)m * PROJW_;
    const float* wr = Wdt + (size_t)d * DR_;
    float acc = bdt[d];
    #pragma unroll
    for (int r = 0; r < DR_; ++r) acc += pr[r] * wr[r];
    dt[idx] = (acc > 20.0f) ? acc : log1pf(__expf(acc));
}

// ---------------------------------------------------------------------------
// Chunked parallel scan.  Recurrence h[t] = a[t]*h[t-1] + u[t] is elementwise
// linear -> per-chunk pairs (prod a, b) compose associatively.
// lane = s + 16*(d&1); wave handles 2 channels x 16 states.
//
// Pass A: per-(b,chunk,d,s) compute (aP, bP) with h_in = 0.
// ---------------------------------------------------------------------------
__global__ __launch_bounds__(256) void scan_partial_kernel(
    const float* __restrict__ dt, const float* __restrict__ proj,
    const float* __restrict__ xh, const float* __restrict__ A_log,
    float* __restrict__ chA, float* __restrict__ chB)
{
    int b    = blockIdx.z;
    int ck   = blockIdx.y;
    int lane = threadIdx.x & 31;
    int wv   = threadIdx.x >> 5;
    int d    = blockIdx.x * 16 + wv * 2 + (lane >> 4);
    int s    = lane & 15;

    float Av = -__expf(A_log[d * DS_ + s]);
    float aP = 1.0f, h = 0.0f;
    size_t rowd = (size_t)b * L_ * DI_ + d;
    size_t rowp = (size_t)b * L_ * PROJW_;

    int t0 = ck * CH_;
    for (int t = t0; t < t0 + CH_; ++t) {
        float dtv = dt[rowd + (size_t)t * DI_];
        float xv  = xh[rowd + (size_t)t * DI_];
        float Bv  = proj[rowp + (size_t)t * PROJW_ + DR_ + s];
        float dA  = __expf(dtv * Av);
        aP *= dA;
        h = dA * h + (dtv * xv) * Bv;
    }
    size_t o = (((size_t)b * NCH_ + ck) * DI_ + d) * DS_ + s;
    chA[o] = aP;
    chB[o] = h;
}

// Pass B: exclusive scan across chunks per (b,d,s); CHB <- h_in per chunk.
__global__ __launch_bounds__(256) void scan_combine_kernel(
    const float* __restrict__ chA, float* __restrict__ chB)
{
    int idx = blockIdx.x * 256 + threadIdx.x;   // (b*DI + d)*DS + s
    if (idx >= B_ * DI_ * DS_) return;
    int b  = idx / (DI_ * DS_);
    int ds = idx % (DI_ * DS_);
    float h = 0.0f;
    for (int c = 0; c < NCH_; ++c) {
        size_t o = ((size_t)b * NCH_ + c) * (DI_ * DS_) + ds;
        float aP = chA[o];
        float bP = chB[o];
        chB[o] = h;                 // exclusive prefix = chunk's initial state
        h = aP * h + bP;
    }
}

// Pass C: re-run chunk from known h_in, emit y via 16-lane shfl butterfly.
__global__ __launch_bounds__(256) void scan_emit_kernel(
    const float* __restrict__ dt, const float* __restrict__ proj,
    const float* __restrict__ xh, const float* __restrict__ A_log,
    const float* __restrict__ chB, float* __restrict__ y)
{
    int b    = blockIdx.z;
    int ck   = blockIdx.y;
    int lane = threadIdx.x & 31;
    int wv   = threadIdx.x >> 5;
    int d    = blockIdx.x * 16 + wv * 2 + (lane >> 4);
    int s    = lane & 15;

    float Av = -__expf(A_log[d * DS_ + s]);
    float h  = chB[(((size_t)b * NCH_ + ck) * DI_ + d) * DS_ + s];
    size_t rowd = (size_t)b * L_ * DI_ + d;
    size_t rowp = (size_t)b * L_ * PROJW_;

    int t0 = ck * CH_;
    for (int t = t0; t < t0 + CH_; ++t) {
        float dtv = dt[rowd + (size_t)t * DI_];
        float xv  = xh[rowd + (size_t)t * DI_];
        float Bv  = proj[rowp + (size_t)t * PROJW_ + DR_ + s];
        float Cv  = proj[rowp + (size_t)t * PROJW_ + DR_ + DS_ + s];
        float dA  = __expf(dtv * Av);
        h = dA * h + (dtv * xv) * Bv;
        float yp = h * Cv;
        yp += __shfl_xor(yp, 1, 32);
        yp += __shfl_xor(yp, 2, 32);
        yp += __shfl_xor(yp, 4, 32);
        yp += __shfl_xor(yp, 8, 32);
        if (s == 0) y[rowd + (size_t)t * DI_] = yp;
    }
}

// ---------------------------------------------------------------------------
// gate: y2 = (y + xh*D) * silu(z); emit bf16 for GEMM3
// ---------------------------------------------------------------------------
__global__ __launch_bounds__(256) void gate_kernel(
    const float* __restrict__ y, const float* __restrict__ xh,
    const float* __restrict__ xz, const float* __restrict__ Dsk,
    __bf16* __restrict__ out)
{
    int idx = blockIdx.x * 256 + threadIdx.x;
    if (idx >= ML_ * DI_) return;
    int m = idx / DI_, d = idx % DI_;
    float z = xz[(size_t)m * (2 * DI_) + DI_ + d];
    float v = (y[idx] + xh[idx] * Dsk[d]) * silu_f(z);
    out[idx] = (__bf16)v;
}

// ---------------------------------------------------------------------------
// head: out[m] = dot(x[m,:], head_W) + head_b   (one row per wave)
// ---------------------------------------------------------------------------
__global__ __launch_bounds__(256) void head_kernel(
    const float* __restrict__ x, const float* __restrict__ hw,
    const float* __restrict__ hb, float* __restrict__ out)
{
    int wv = threadIdx.x >> 5, lane = threadIdx.x & 31;
    int m = blockIdx.x * 8 + wv;
    const float* xr = x + (size_t)m * DM_;
    float acc = 0.0f;
    #pragma unroll
    for (int dcol = lane; dcol < DM_; dcol += 32) acc += xr[dcol] * hw[dcol];
    acc += __shfl_xor(acc, 16, 32);
    acc += __shfl_xor(acc, 8, 32);
    acc += __shfl_xor(acc, 4, 32);
    acc += __shfl_xor(acc, 2, 32);
    acc += __shfl_xor(acc, 1, 32);
    if (lane == 0) out[m] = acc + hb[0];
}

// ---------------------------------------------------------------------------
extern "C" void kernel_launch(void* const* d_in, const int* in_sizes, int n_in,
                              void* d_out, int out_size, void* d_ws, size_t ws_size,
                              hipStream_t stream)
{
    const float* features  = (const float*)d_in[0];
    const float* embed_W   = (const float*)d_in[1];
    const float* embed_b   = (const float*)d_in[2];
    const float* in_proj_W = (const float*)d_in[3];
    const float* conv_W    = (const float*)d_in[4];
    const float* conv_b    = (const float*)d_in[5];
    const float* x_proj_W  = (const float*)d_in[6];
    const float* dt_proj_W = (const float*)d_in[7];
    const float* dt_proj_b = (const float*)d_in[8];
    const float* A_log     = (const float*)d_in[9];
    const float* D_skip    = (const float*)d_in[10];
    const float* out_proj_W= (const float*)d_in[11];
    const float* head_W    = (const float*)d_in[12];
    const float* head_b    = (const float*)d_in[13];

    char* ws = (char*)d_ws;
    const size_t MB = (size_t)1 << 20;
    float*  X[2] = { (float*)(ws + 0 * MB), (float*)(ws + 8 * MB) };
    float*  XZ   = (float*)(ws + 16 * MB);   // (ML, 1024)  32 MB
    float*  XH   = (float*)(ws + 48 * MB);   // (ML, 512)   16 MB
    float*  PROJ = (float*)(ws + 64 * MB);   // (ML, 48)    1.5 MB
    float*  DT   = (float*)(ws + 66 * MB);   // (ML, 512)   16 MB
    float*  Y    = (float*)(ws + 82 * MB);   // (ML, 512)   16 MB
    __bf16* ABF  = (__bf16*)(ws + 98 * MB);  // bf16 activation staging (<=8MB)
    __bf16* WBF  = (__bf16*)(ws + 106 * MB); // bf16 weight staging (<=512KB)
    float*  CHA  = (float*)(ws + 107 * MB);  // (B,NCH,DI,DS) chunk prodA  4 MB
    float*  CHB  = (float*)(ws + 111 * MB);  // (B,NCH,DI,DS) chunk b / h0 4 MB

    embed_kernel<<<(ML_ * DM_ + 255) / 256, 256, 0, stream>>>(
        features, embed_W, embed_b, X[0]);

    int cur = 0;
    for (int i = 0; i < NL_; ++i) {
        const float* Wi   = in_proj_W  + (size_t)i * 2 * DI_ * DM_;
        const float* Wci  = conv_W     + (size_t)i * DI_ * DC_;
        const float* bci  = conv_b     + (size_t)i * DI_;
        const float* Wxi  = x_proj_W   + (size_t)i * PROJW_ * DI_;
        const float* Wdti = dt_proj_W  + (size_t)i * DI_ * DR_;
        const float* bdti = dt_proj_b  + (size_t)i * DI_;
        const float* Ali  = A_log      + (size_t)i * DI_ * DS_;
        const float* Dski = D_skip     + (size_t)i * DI_;
        const float* Woi  = out_proj_W + (size_t)i * DM_ * DI_;

        // GEMM1: xz = x @ W_in^T
        cvt_kernel<<<(ML_ * DM_ + 255) / 256, 256, 0, stream>>>(X[cur], ABF, ML_ * DM_);
        cvt_kernel<<<(2 * DI_ * DM_ + 255) / 256, 256, 0, stream>>>(Wi, WBF, 2 * DI_ * DM_);
        gemm_wmma_bf16<<<dim3((2 * DI_) / 128, ML_ / 128), 256, 0, stream>>>(
            ABF, WBF, XZ, ML_, 2 * DI_, DM_);

        conv_silu_kernel<<<(ML_ * DI_ + 255) / 256, 256, 0, stream>>>(XZ, Wci, bci, XH);

        // GEMM2: proj = xh @ Wx^T   (N = 48, masked)
        cvt_kernel<<<(ML_ * DI_ + 255) / 256, 256, 0, stream>>>(XH, ABF, ML_ * DI_);
        cvt_kernel<<<(PROJW_ * DI_ + 255) / 256, 256, 0, stream>>>(Wxi, WBF, PROJW_ * DI_);
        gemm_wmma_bf16<<<dim3(1, ML_ / 128), 256, 0, stream>>>(
            ABF, WBF, PROJ, ML_, PROJW_, DI_);

        dt_kernel<<<(ML_ * DI_ + 255) / 256, 256, 0, stream>>>(PROJ, Wdti, bdti, DT);

        // chunked parallel scan
        scan_partial_kernel<<<dim3(DI_ / 16, NCH_, B_), 256, 0, stream>>>(
            DT, PROJ, XH, Ali, CHA, CHB);
        scan_combine_kernel<<<(B_ * DI_ * DS_ + 255) / 256, 256, 0, stream>>>(CHA, CHB);
        scan_emit_kernel<<<dim3(DI_ / 16, NCH_, B_), 256, 0, stream>>>(
            DT, PROJ, XH, Ali, CHB, Y);

        gate_kernel<<<(ML_ * DI_ + 255) / 256, 256, 0, stream>>>(Y, XH, XZ, Dski, ABF);

        // GEMM3: x' = y2 @ W_out^T
        cvt_kernel<<<(DM_ * DI_ + 255) / 256, 256, 0, stream>>>(Woi, WBF, DM_ * DI_);
        gemm_wmma_bf16<<<dim3(DM_ / 128, ML_ / 128), 256, 0, stream>>>(
            ABF, WBF, X[cur ^ 1], ML_, DM_, DI_);
        cur ^= 1;
    }

    head_kernel<<<ML_ / 8, 256, 0, stream>>>(X[cur], head_W, head_b, (float*)d_out);
}
